// BiLSTM_CRF_30571577213094
// MI455X (gfx1250) — matrix operand order repaired and checked by reading, not compiled, not run on hardware
//
#include <hip/hip_runtime.h>
#include <hip/hip_bf16.h>
#include <math.h>

// ---------------------------------------------------------------------------
// BiLSTM-CRF for MI455X (gfx1250, wave32, v_wmma_f32_16x16x32_f16)
//
//  0) cast_emb_kernel : emb f32 -> f16 once (kills per-tile cvt storm)
//  1) pack_b_kernel   : W ([N][K] f32) -> WMMA B-fragment layout f16
//  2) xproj_kernel    : per 16-row block, stage A-tile in LDS via
//                       global_load_async_to_lds_b128 (ASYNCcnt), 8 waves
//                       sweep all 64 n-tiles; output written directly in the
//                       LSTM C-fragment layout [dir][t][mt][jt][gate][lane][8]
//  3) lstm_kernel     : 1 WG (32 waves)/direction; h (f16) in LDS (padded
//                       rows, conflict-free); c in registers in C layout;
//                       Whh fragments streamed from L2; contiguous b128
//                       xproj fragment reads per step
//  4) feats_kernel    : [B*T,512] @ WtagT (N padded 12->16), WMMA
//  5) viterbi_kernel  : K=12 forward + backtrace, one wave per batch row
// ---------------------------------------------------------------------------

#define Bx 64
#define Tx 1024
#define Dx 256
#define HHx 256
#define G4 1024   /* 4*Hh */
#define Kt 12
#define NEGV (-10000.0f)
#define HPAD 264  /* padded LDS row stride in halves (bank-conflict free) */

typedef __attribute__((ext_vector_type(16))) _Float16 v16h;
typedef __attribute__((ext_vector_type(8)))  _Float16 v8h;
typedef __attribute__((ext_vector_type(8)))  float    v8f;
typedef __attribute__((ext_vector_type(4)))  int      v4i;

__device__ __forceinline__ v8f wmma16(v16h a, v16h b, v8f c) {
  return __builtin_amdgcn_wmma_f32_16x16x32_f16(false, a, false, b, (short)0, c,
                                                false, false);
}

__device__ __forceinline__ v16h joinh(v8h lo, v8h hi) {
  v16h r;
#pragma unroll
  for (int i = 0; i < 8; ++i) { r[i] = lo[i]; r[8 + i] = hi[i]; }
  return r;
}

__device__ __forceinline__ float sigf(float x) {
  return 1.0f / (1.0f + __expf(-x));
}

// --- async global->LDS 16B copy (CDNA5 ASYNCcnt path), safe fallback -------
#if defined(__HIP_DEVICE_COMPILE__) && \
    __has_builtin(__builtin_amdgcn_global_load_async_to_lds_b128) && \
    __has_builtin(__builtin_amdgcn_s_wait_asynccnt)
#define HAVE_ASYNC_LDS 1
#else
#define HAVE_ASYNC_LDS 0
#endif

__device__ __forceinline__ void async_copy16(const _Float16* g, _Float16* l) {
#if HAVE_ASYNC_LDS
  // prototype (from hipcc diagnostic): (AS1 v4i*, AS3 v4i*, imm off, imm cpol)
  __attribute__((address_space(1))) v4i* gp =
      (__attribute__((address_space(1))) v4i*)
      (__attribute__((address_space(1))) void*)g;
  __attribute__((address_space(3))) v4i* lp =
      (__attribute__((address_space(3))) v4i*)
      (__attribute__((address_space(3))) void*)l;
  __builtin_amdgcn_global_load_async_to_lds_b128(gp, lp, 0, 0);
#else
  *(v8h*)l = *(const v8h*)g;
#endif
}

__device__ __forceinline__ void wait_async() {
#if HAVE_ASYNC_LDS
  __builtin_amdgcn_s_wait_asynccnt(0);
#endif
}

// ---------------------------------------------------------------------------
// 0) emb f32 -> f16 (one pass; removes redundant conversions from the GEMM)
// ---------------------------------------------------------------------------
__global__ void cast_emb_kernel(const float* __restrict__ emb,
                                _Float16* __restrict__ emb_h, int total) {
  int idx = blockIdx.x * blockDim.x + threadIdx.x;
  if (idx < total) emb_h[idx] = (_Float16)emb[idx];
}

// ---------------------------------------------------------------------------
// 1) Repack W (row-major [N][K], f32) into B-fragment layout:
//   out[((nt*KT + kt)*32 + lane)*16 + i] = W[n][k],
//   n = nt*16 + (lane&15), k = kt*32 + (lane>>4)*16 + i   (zero-pad n >= N)
// ---------------------------------------------------------------------------
__global__ void pack_b_kernel(const float* __restrict__ W,
                              _Float16* __restrict__ out,
                              int N, int K, int NT) {
  int idx = blockIdx.x * blockDim.x + threadIdx.x;
  int KT = K >> 5;
  int total = NT * KT * 512;
  if (idx >= total) return;
  int i    = idx & 15;
  int lane = (idx >> 4) & 31;
  int kt   = (idx >> 9) % KT;
  int nt   = idx / (KT * 512);
  int k = kt * 32 + ((lane >> 4) << 4) + i;
  int n = nt * 16 + (lane & 15);
  out[idx] = (n < N) ? (_Float16)W[(size_t)n * K + k] : (_Float16)0.0f;
}

// ---------------------------------------------------------------------------
// 2) xproj: grid (4096, 2), block 256 (8 waves).
//    Stage 16x256 f16 A-tile in LDS (async), sweep all 64 n-tiles.
//    Output layout (LSTM fragment order):
//      xproj_l[((((dir*T + t)*4 + mt_l)*16 + jt)*4 + gate)*256 + lane_l*8 + r_l]
//      with b = row>>10, t = row&1023, mt_l=b>>4, lane_l=((b>>3)&1)*16+(n&15),
//      r_l=b&7, gate=n>>8, jt=(n>>4)&15
// ---------------------------------------------------------------------------
__global__ __launch_bounds__(256) void xproj_kernel(
    const int* __restrict__ sentence, const _Float16* __restrict__ emb_h,
    const _Float16* __restrict__ wih_pack,   // [2][64][8][32][16]
    const float* __restrict__ bih_f, const float* __restrict__ bhh_f,
    const float* __restrict__ bih_b, const float* __restrict__ bhh_b,
    _Float16* __restrict__ xproj_l)
{
  __shared__ _Float16 xsh[16 * HPAD];        // padded A tile (~8.4 KB)

  const int tid = threadIdx.x;
  const int mt  = blockIdx.x;                // 0..4095 over B*T rows
  const int dir = blockIdx.y;

  // stage A tile: 16 rows x 512B, 16B chunks, 2 per thread (async to LDS)
  for (int c = tid; c < 512; c += 256) {
    int rrow = c >> 5;                       // 0..15
    int off  = (c & 31) * 8;                 // halves
    int tok  = sentence[mt * 16 + rrow];
    async_copy16(emb_h + (size_t)tok * Dx + off, &xsh[rrow * HPAD + off]);
  }
  wait_async();
  __syncthreads();

  const int wave = tid >> 5, lane = tid & 31;
  const int half = lane >> 4, ncol = lane & 15;
  const _Float16* wpd = wih_pack + (size_t)dir * 262144;
  const float* bih = dir ? bih_b : bih_f;
  const float* bhh = dir ? bhh_b : bhh_f;

#pragma unroll
  for (int ni = 0; ni < 8; ++ni) {
    const int nt = wave * 8 + ni;            // 0..63
    v8f acc = {};
#pragma unroll
    for (int kt = 0; kt < 8; ++kt) {
      const int kb = half * 16;
      v16h a = joinh(*(const v8h*)&xsh[ncol * HPAD + kt * 32 + kb],
                     *(const v8h*)&xsh[ncol * HPAD + kt * 32 + kb + 8]);
      const _Float16* bp = wpd + (size_t)nt * 4096 + ((size_t)kt * 32 + lane) * 16;
      v16h bf = joinh(*(const v8h*)bp, *(const v8h*)(bp + 8));
      acc = wmma16(a, bf, acc);
    }
    const int n = nt * 16 + ncol;
    const float bias = bih[n] + bhh[n];
    const int gate = n >> 8;
    const int jt   = (n >> 4) & 15;
#pragma unroll
    for (int r = 0; r < 8; ++r) {
      int row = mt * 16 + r + 8 * half;
      int b = row >> 10, t = row & (Tx - 1);
      int lane_l = ((b >> 3) & 1) * 16 + ncol;
      int r_l = b & 7;
      int mt_l = b >> 4;
      size_t idx = (((((size_t)dir * Tx + t) * 4 + mt_l) * 16 + jt) * 4 + gate);
      xproj_l[idx * 256 + lane_l * 8 + r_l] = (_Float16)(acc[r] + bias);
    }
  }
}

// ---------------------------------------------------------------------------
// 3) LSTM recurrence: 1 WG (1024 thr = 32 waves) per direction.
// ---------------------------------------------------------------------------
__global__ __launch_bounds__(1024) void lstm_kernel(
    const _Float16* __restrict__ xproj_l,   // fragment layout (see above)
    const _Float16* __restrict__ whh_pack,  // [2][64][8][32][16]
    const int* __restrict__ seq_lens,
    const float* __restrict__ h0,           // [2][B][Hh]
    const float* __restrict__ c0,           // [2][B][Hh]
    _Float16* __restrict__ hcat)            // [B*T][2*Hh]
{
  __shared__ _Float16 hsh[Bx * HPAD];       // ~33 KB, padded rows

  const int dir  = blockIdx.x;
  const int tid  = threadIdx.x;
  const int lane = tid & 31;
  const int wave = tid >> 5;
  const int half = lane >> 4;
  const int ncol = lane & 15;

  for (int i = tid; i < Bx * HHx; i += 1024) {
    int b = i >> 8, j = i & 255;
    hsh[b * HPAD + j] = (_Float16)h0[(size_t)dir * Bx * HHx + i];
  }

  v8f cfr[2];
  int mbase[2], jt_[2];
  int sl[2][8];
#pragma unroll
  for (int ti = 0; ti < 2; ++ti) {
    int task = wave * 2 + ti;
    int mt = task >> 4;      // 0..3
    int jt = task & 15;      // 0..15
    mbase[ti] = mt * 16;
    jt_[ti] = jt;
    int j = jt * 16 + ncol;
#pragma unroll
    for (int r = 0; r < 8; ++r) {
      int b = mt * 16 + r + 8 * half;
      cfr[ti][r] = c0[(size_t)dir * Bx * HHx + (size_t)b * HHx + j];
      sl[ti][r]  = seq_lens[b];
    }
  }
  __syncthreads();

  const _Float16* wp_base = whh_pack + (size_t)dir * 262144;

  v8f hn[2];
  for (int step = 0; step < Tx; ++step) {
    const int t = dir ? (Tx - 1 - step) : step;

#pragma unroll
    for (int ti = 0; ti < 2; ++ti) {
      const int jt = jt_[ti];
      v8f acc0 = {}, acc1 = {}, acc2 = {}, acc3 = {};
#pragma unroll
      for (int kt = 0; kt < 8; ++kt) {
        const int m  = mbase[ti] + ncol;
        const int kb = half * 16;
        v16h a = joinh(*(const v8h*)&hsh[m * HPAD + kt * 32 + kb],
                       *(const v8h*)&hsh[m * HPAD + kt * 32 + kb + 8]);
        const _Float16* bp;
        v16h bf;
        bp = wp_base + ((size_t)((0 * 16 + jt) * 8 + kt) * 32 + lane) * 16;
        bf = joinh(*(const v8h*)bp, *(const v8h*)(bp + 8));
        acc0 = wmma16(a, bf, acc0);
        bp = wp_base + ((size_t)((1 * 16 + jt) * 8 + kt) * 32 + lane) * 16;
        bf = joinh(*(const v8h*)bp, *(const v8h*)(bp + 8));
        acc1 = wmma16(a, bf, acc1);
        bp = wp_base + ((size_t)((2 * 16 + jt) * 8 + kt) * 32 + lane) * 16;
        bf = joinh(*(const v8h*)bp, *(const v8h*)(bp + 8));
        acc2 = wmma16(a, bf, acc2);
        bp = wp_base + ((size_t)((3 * 16 + jt) * 8 + kt) * 32 + lane) * 16;
        bf = joinh(*(const v8h*)bp, *(const v8h*)(bp + 8));
        acc3 = wmma16(a, bf, acc3);
      }
      // xproj fragments: 4 contiguous 16B loads per task
      const size_t tb =
          ((((size_t)dir * Tx + t) * 4 + (mbase[ti] >> 4)) * 16 + jt) * 4;
      v8h x0 = *(const v8h*)&xproj_l[(tb + 0) * 256 + lane * 8];
      v8h x1 = *(const v8h*)&xproj_l[(tb + 1) * 256 + lane * 8];
      v8h x2 = *(const v8h*)&xproj_l[(tb + 2) * 256 + lane * 8];
      v8h x3 = *(const v8h*)&xproj_l[(tb + 3) * 256 + lane * 8];

      // prefetch next step's fragment block for this task
      if (step + 1 < Tx) {
        int tn = dir ? (Tx - 2 - step) : (step + 1);
        size_t tbn =
            ((((size_t)dir * Tx + tn) * 4 + (mbase[ti] >> 4)) * 16 + jt) * 4;
        __builtin_prefetch(&xproj_l[tbn * 256], 0, 1);
      }

      const int j = jt * 16 + ncol;
#pragma unroll
      for (int r = 0; r < 8; ++r) {
        int b = mbase[ti] + r + 8 * half;
        float gi = acc0[r] + (float)x0[r];
        float gf = acc1[r] + (float)x1[r];
        float gg = acc2[r] + (float)x2[r];
        float go = acc3[r] + (float)x3[r];
        float c2 = sigf(gf) * cfr[ti][r] + sigf(gi) * tanhf(gg);
        float h2 = sigf(go) * tanhf(c2);
        bool  mk = (t < sl[ti][r]);
        float hold = (float)hsh[b * HPAD + j];
        hn[ti][r]  = mk ? h2 : hold;
        cfr[ti][r] = mk ? c2 : cfr[ti][r];
      }
    }
    __syncthreads();  // reads of hsh complete

#pragma unroll
    for (int ti = 0; ti < 2; ++ti) {
      const int j = jt_[ti] * 16 + ncol;
#pragma unroll
      for (int r = 0; r < 8; ++r) {
        int b = mbase[ti] + r + 8 * half;
        _Float16 hv = (_Float16)hn[ti][r];
        hsh[b * HPAD + j] = hv;
        hcat[((size_t)b * Tx + t) * (size_t)(2 * HHx) + dir * HHx + j] = hv;
      }
    }
    __syncthreads();  // writes visible before next step
  }
}

// ---------------------------------------------------------------------------
// 4) feats[row][k] = (hcat[row] @ Wtag^T)[k] + btag[k]   (N padded to 16)
// ---------------------------------------------------------------------------
__global__ __launch_bounds__(256) void feats_kernel(
    const _Float16* __restrict__ hcat,       // [B*T][512]
    const _Float16* __restrict__ wtag_pack,  // [16][32][16]
    const float* __restrict__ btag,
    float* __restrict__ feats)               // [B*T][12]
{
  const int wave = threadIdx.x >> 5;
  const int lane = threadIdx.x & 31;
  const int half = lane >> 4;
  const int ncol = lane & 15;
  const int mt = blockIdx.x * 8 + wave;      // 0..4095
  const int m  = mt * 16 + ncol;

  v8f acc = {};
#pragma unroll
  for (int kt = 0; kt < 16; ++kt) {
    v16h a = joinh(*(const v8h*)&hcat[(size_t)m * 512 + kt * 32 + half * 16],
                   *(const v8h*)&hcat[(size_t)m * 512 + kt * 32 + half * 16 + 8]);
    const _Float16* bp = wtag_pack + ((size_t)kt * 32 + lane) * 16;
    v16h bf = joinh(*(const v8h*)bp, *(const v8h*)(bp + 8));
    acc = wmma16(a, bf, acc);
  }
  if (ncol < Kt) {
    float bt = btag[ncol];
#pragma unroll
    for (int r = 0; r < 8; ++r) {
      int row = mt * 16 + r + 8 * half;
      feats[(size_t)row * Kt + ncol] = acc[r] + bt;
    }
  }
}

// ---------------------------------------------------------------------------
// 5) Viterbi: one wave per batch row, shuffles for the 12-wide max.
// ---------------------------------------------------------------------------
__global__ __launch_bounds__(32) void viterbi_kernel(
    const float* __restrict__ feats, const float* __restrict__ trans,
    const int* __restrict__ seq_lens, int* __restrict__ bptrs,
    float* __restrict__ out)
{
  const int b = blockIdx.x;
  const int lane = threadIdx.x;
  const bool act = lane < Kt;
  const int len = seq_lens[b];

  float tr[Kt];
#pragma unroll
  for (int k = 0; k < Kt; ++k)
    tr[k] = act ? trans[lane * Kt + k] : -3.0e38f;

  float fv = (lane == 10) ? 0.0f : NEGV;  // START = 10

  for (int t = 0; t < Tx; ++t) {
    float best = -3.0e38f;
    int bp = 0;
#pragma unroll
    for (int k = 0; k < Kt; ++k) {
      float fk = __shfl(fv, k, 32);
      float s = fk + tr[k];
      if (s > best) { best = s; bp = k; }
    }
    float ft = act ? feats[((size_t)b * Tx + t) * Kt + lane] : 0.0f;
    bool  mk = (t < len);
    fv = (mk && act) ? (best + ft) : fv;
    int bpw = mk ? bp : lane;               // identity passthrough on pad
    if (act) bptrs[((size_t)b * Tx + t) * Kt + lane] = bpw;
  }

  float term = act ? (fv + trans[11 * Kt + lane]) : -3.0e38f;  // STOP = 11
  float bv = -3.0e38f;
  int bk = 0;
#pragma unroll
  for (int k = 0; k < Kt; ++k) {
    float tk = __shfl(term, k, 32);
    if (tk > bv) { bv = tk; bk = k; }
  }

  if (lane == 0) {
    out[b] = bv;
    int tag = bk;
    for (int t = Tx - 1; t >= 0; --t) {
      out[64 + (size_t)b * Tx + t] = (t < len) ? (float)tag : 0.0f;
      tag = bptrs[((size_t)b * Tx + t) * Kt + tag];
    }
  }
}

// ---------------------------------------------------------------------------
extern "C" void kernel_launch(void* const* d_in, const int* in_sizes, int n_in,
                              void* d_out, int out_size, void* d_ws,
                              size_t ws_size, hipStream_t stream) {
  (void)in_sizes; (void)n_in; (void)out_size; (void)ws_size;

  const int*   sentence = (const int*)  d_in[0];
  const int*   seq_lens = (const int*)  d_in[1];
  const float* emb      = (const float*)d_in[2];
  const float* Wih_f    = (const float*)d_in[3];
  const float* Whh_f    = (const float*)d_in[4];
  const float* bih_f    = (const float*)d_in[5];
  const float* bhh_f    = (const float*)d_in[6];
  const float* Wih_b    = (const float*)d_in[7];
  const float* Whh_b    = (const float*)d_in[8];
  const float* bih_b    = (const float*)d_in[9];
  const float* bhh_b    = (const float*)d_in[10];
  const float* Wtag     = (const float*)d_in[11];
  const float* btag     = (const float*)d_in[12];
  const float* trans    = (const float*)d_in[13];
  const float* h0       = (const float*)d_in[14];
  const float* c0       = (const float*)d_in[15];
  float* out = (float*)d_out;

  char* ws = (char*)d_ws;
  _Float16* whh_pack  = (_Float16*)(ws);                      //   1 MB
  _Float16* wih_pack  = (_Float16*)(ws + (1ull   << 20));     //   1 MB
  _Float16* wtag_pack = (_Float16*)(ws + (2ull   << 20));     //  16 KB
  _Float16* emb_h     = (_Float16*)(ws + (3ull   << 20));     // ~25 MB
  _Float16* xproj_l   = (_Float16*)(ws + (32ull  << 20));     // 256 MB
  _Float16* hcat      = (_Float16*)(ws + (288ull << 20));     //  64 MB
  int*      bptrs     = (int*)     (ws + (352ull << 20));     //   3 MB
  float*    feats     = (float*)   (ws + (356ull << 20));     //   3 MB

  // 0) emb -> f16
  const int embN = 50000 * Dx;
  cast_emb_kernel<<<(embN + 255) / 256, 256, 0, stream>>>(emb, emb_h, embN);

  // 1) repack weights into WMMA B-fragment layout (f16)
  pack_b_kernel<<<1024, 256, 0, stream>>>(Wih_f, wih_pack,          G4, Dx, 64);
  pack_b_kernel<<<1024, 256, 0, stream>>>(Wih_b, wih_pack + 262144, G4, Dx, 64);
  pack_b_kernel<<<1024, 256, 0, stream>>>(Whh_f, whh_pack,          G4, HHx, 64);
  pack_b_kernel<<<1024, 256, 0, stream>>>(Whh_b, whh_pack + 262144, G4, HHx, 64);
  pack_b_kernel<<<32,   256, 0, stream>>>(Wtag,  wtag_pack, Kt, 2 * HHx, 1);

  // 2) input projection (async LDS staging + WMMA, fragment-layout output)
  xproj_kernel<<<dim3(4096, 2), 256, 0, stream>>>(
      sentence, emb_h, wih_pack, bih_f, bhh_f, bih_b, bhh_b, xproj_l);

  // 3) bidirectional LSTM recurrence (WMMA, h in LDS)
  lstm_kernel<<<2, 1024, 0, stream>>>(xproj_l, whh_pack, seq_lens, h0, c0, hcat);

  // 4) tag projection (WMMA, N padded to 16)
  feats_kernel<<<512, 256, 0, stream>>>(hcat, wtag_pack, btag, feats);

  // 5) CRF Viterbi decode + backtrace
  viterbi_kernel<<<64, 32, 0, stream>>>(feats, trans, seq_lens, bptrs, out);
}
